// Decoder_48146583388405
// MI455X (gfx1250) — matrix operand order repaired and checked by reading, not compiled, not run on hardware
//
#include <hip/hip_runtime.h>

#define B_ 64
#define T_ 256
#define U_ 1024
#define V_ 512
#define L_ 512
#define F_ 64
#define D0_ (L_ + F_)   /* 576 */
#define G4U (4 * U_)    /* 4096 */

typedef __bf16 v16bf __attribute__((ext_vector_type(16)));
typedef __bf16 v8bf  __attribute__((ext_vector_type(8)));
typedef float  v8f   __attribute__((ext_vector_type(8)));

__device__ __forceinline__ unsigned short f2bf(float f) {
  unsigned int u = __float_as_uint(f);
  u = u + 0x7FFFu + ((u >> 16) & 1u);   // round-to-nearest-even
  return (unsigned short)(u >> 16);
}

// ---------------- prep kernels ----------------

// dst[n*D + k] = bf16( k < K0 ? kx[k*N + n] : rk[(k-K0)*N + n] ), D = K0+K1
__global__ void prep_weights(const float* __restrict__ kx, const float* __restrict__ rk,
                             unsigned short* __restrict__ dst, int K0, int K1, int N) {
  const int D = K0 + K1;
  const long total = (long)N * D;
  long idx = (long)blockIdx.x * blockDim.x + threadIdx.x;
  if (idx >= total) return;
  int n = (int)(idx / D);
  int k = (int)(idx % D);
  float v = (k < K0) ? kx[(long)k * N + n] : rk[(long)(k - K0) * N + n];
  dst[idx] = f2bf(v);
}

// inp[b,t,k] = bf16( k < L ? latent[b,k] : x[b,t,k-L] )
__global__ void prep_concat(const float* __restrict__ latent, const float* __restrict__ x,
                            unsigned short* __restrict__ inp) {
  const long total = (long)B_ * T_ * D0_;
  long idx = (long)blockIdx.x * blockDim.x + threadIdx.x;
  if (idx >= total) return;
  int k = (int)(idx % D0_);
  long bt = idx / D0_;
  int t = (int)(bt % T_);
  int b = (int)(bt / T_);
  float v = (k < L_) ? latent[(long)b * L_ + k]
                     : x[((long)b * T_ + t) * F_ + (k - L_)];
  inp[idx] = f2bf(v);
}

// zero 6 h buffers (bf16) and 3 c buffers (f32)
__global__ void prep_init(unsigned short* __restrict__ h, float* __restrict__ c) {
  long idx = (long)blockIdx.x * blockDim.x + threadIdx.x;
  if (idx < (long)6 * B_ * U_) h[idx] = 0;
  if (idx < (long)3 * B_ * U_) c[idx] = 0.f;
}

// ---------------- WMMA GEMM core ----------------
// One wave accumulates ONE 16(M)x16(N) tile: A row-major bf16 (stride sA halves),
// B row = one weight row (N-major fused layout), contiguous K.
__device__ __forceinline__ v8f mma_run(v8f acc,
    const unsigned short* __restrict__ arow,
    const unsigned short* __restrict__ brow,
    int Klen, int half) {
  for (int kc = 0; kc < Klen; kc += 32) {
    // 16-bit A-matrix 16x32 per-lane layout (two 16B chunks)
    v8bf alo = *(const v8bf*)(arow + kc + half * 8);
    v8bf ahi = *(const v8bf*)(arow + kc + 16 + half * 8);
    v16bf a;
#pragma unroll
    for (int j = 0; j < 8; ++j) { a[j] = alo[j]; a[8 + j] = ahi[j]; }
    __builtin_prefetch(brow + kc + 128, 0, 0);   // global_prefetch_b8
    // 16-bit B-matrix 32x16: same two 16B K chunks as A
    v8bf blo = *(const v8bf*)(brow + kc + half * 8);
    v8bf bhi = *(const v8bf*)(brow + kc + 16 + half * 8);
    v16bf b;
#pragma unroll
    for (int j = 0; j < 8; ++j) { b[j] = blo[j]; b[8 + j] = bhi[j]; }
    acc = __builtin_amdgcn_wmma_f32_16x16x32_bf16(
        false, a, false, b, (short)0, acc, false, false);
  }
  return acc;
}

// One timestep of one LSTM layer.
// grid = U/16 = 64 blocks, block = 512 threads = 16 waves (4 m-tiles x 4 gates).
// Each wave owns ONE gate tile -> 4x shorter dependent WMMA chain, 4 waves/SIMD.
// Gate tiles meet in LDS for the fused elementwise cell update.
__global__ void __launch_bounds__(512)
lstm_cell(const unsigned short* __restrict__ A0, long sA0, int K0,
          const unsigned short* __restrict__ Ar,
          const unsigned short* __restrict__ W, int D,
          const float* __restrict__ bias,
          float* __restrict__ c,
          unsigned short* __restrict__ hout,
          unsigned short* __restrict__ yout, long sY) {
  __shared__ float zbuf[4][B_][16];   // [gate][m][u_local] = 16 KB

  const int lane = threadIdx.x & 31;
  const int wid  = threadIdx.x >> 5;   // 0..15
  const int mw   = wid & 3;            // m-tile (16 rows each)
  const int g    = wid >> 2;           // gate: 0=i 1=f 2=g 3=o
  const int m0   = mw * 16;
  const int ucol = blockIdx.x * 16;
  const int half = lane >> 4;
  const int nl   = lane & 15;

  v8f acc = {0.f, 0.f, 0.f, 0.f, 0.f, 0.f, 0.f, 0.f};

  const unsigned short* wrow = W + (long)(g * U_ + ucol + nl) * D;
  // input contribution
  acc = mma_run(acc, A0 + (long)(m0 + nl) * sA0, wrow, K0, half);
  // recurrent contribution
  acc = mma_run(acc, Ar + (long)(m0 + nl) * U_, wrow + K0, U_, half);

  // C/D layout: VGPR r -> M = r + 8*half, N = nl
#pragma unroll
  for (int r = 0; r < 8; ++r) zbuf[g][m0 + r + half * 8][nl] = acc[r];
  __syncthreads();

  // fused elementwise LSTM cell update: 64x16 outputs, 512 threads -> 2 each
#pragma unroll
  for (int e = threadIdx.x; e < B_ * 16; e += 512) {
    const int m = e >> 4;
    const int ul = e & 15;
    const int u = ucol + ul;
    const float iv = 1.f / (1.f + __expf(-(zbuf[0][m][ul] + bias[u])));
    const float fv = 1.f / (1.f + __expf(-(zbuf[1][m][ul] + bias[U_ + u])));
    const float gv = tanhf(zbuf[2][m][ul] + bias[2 * U_ + u]);
    const float ov = 1.f / (1.f + __expf(-(zbuf[3][m][ul] + bias[3 * U_ + u])));
    const long idx = (long)m * U_ + u;
    const float cn = fv * c[idx] + iv * gv;
    const float hn = ov * tanhf(cn);
    c[idx] = cn;
    const unsigned short hb = f2bf(hn);
    hout[idx] = hb;
    if (yout) yout[(long)m * sY + u] = hb;
  }
}

// logits[BT, V] = Y @ w + b. grid = (BT/64, V/32), block = 256 (8 waves: 4 M x 2 N)
__global__ void __launch_bounds__(256)
proj_logits(const unsigned short* __restrict__ Y,
            const unsigned short* __restrict__ WT,
            const float* __restrict__ bias,
            float* __restrict__ logits) {
  const int lane = threadIdx.x & 31;
  const int wid  = threadIdx.x >> 5;   // 0..7
  const int mw = wid >> 1, nw = wid & 1;
  const int m0 = blockIdx.x * 64 + mw * 16;
  const int v0 = blockIdx.y * 32 + nw * 16;
  const int half = lane >> 4, nl = lane & 15;

  v8f acc = {0.f, 0.f, 0.f, 0.f, 0.f, 0.f, 0.f, 0.f};
  acc = mma_run(acc, Y + (long)(m0 + nl) * U_, WT + (long)(v0 + nl) * U_, U_, half);

  const int v = v0 + nl;
  const float bb = bias[v];
#pragma unroll
  for (int r = 0; r < 8; ++r) {
    const int m = m0 + r + half * 8;
    logits[(long)m * V_ + v] = acc[r] + bb;
  }
}

// softmax over V=512 per row; grid = BT rows, block = 256 (2 elems/thread)
__global__ void __launch_bounds__(256)
softmax512(const float* __restrict__ logits, float* __restrict__ yout) {
  __shared__ float red[256];
  const int row = blockIdx.x;
  const int tid = threadIdx.x;
  const float* lr = logits + (long)row * V_;
  float a = lr[tid], b = lr[tid + 256];
  red[tid] = fmaxf(a, b);
  __syncthreads();
  for (int s = 128; s > 0; s >>= 1) {
    if (tid < s) red[tid] = fmaxf(red[tid], red[tid + s]);
    __syncthreads();
  }
  const float mx = red[0];
  __syncthreads();
  float ea = __expf(a - mx), eb = __expf(b - mx);
  red[tid] = ea + eb;
  __syncthreads();
  for (int s = 128; s > 0; s >>= 1) {
    if (tid < s) red[tid] += red[tid + s];
    __syncthreads();
  }
  const float inv = 1.f / red[0];
  yout[(long)row * V_ + tid] = ea * inv;
  yout[(long)row * V_ + tid + 256] = eb * inv;
}

// ---------------- host launch ----------------
extern "C" void kernel_launch(void* const* d_in, const int* in_sizes, int n_in,
                              void* d_out, int out_size, void* d_ws, size_t ws_size,
                              hipStream_t stream) {
  const float* latent = (const float*)d_in[0];
  const float* x   = (const float*)d_in[1];
  const float* k1  = (const float*)d_in[2];
  const float* rk1 = (const float*)d_in[3];
  const float* b1  = (const float*)d_in[4];
  const float* k2  = (const float*)d_in[5];
  const float* rk2 = (const float*)d_in[6];
  const float* b2  = (const float*)d_in[7];
  const float* k3  = (const float*)d_in[8];
  const float* rk3 = (const float*)d_in[9];
  const float* b3  = (const float*)d_in[10];
  const float* w   = (const float*)d_in[11];
  const float* bb  = (const float*)d_in[12];

  char* ws = (char*)d_ws;
  size_t off = 0;
  auto alloc = [&](size_t bytes) -> char* {
    off = (off + 255) & ~(size_t)255;
    char* p = ws + off;
    off += bytes;
    return p;
  };

  unsigned short* W1  = (unsigned short*)alloc((size_t)G4U * (D0_ + U_) * 2);
  unsigned short* W2  = (unsigned short*)alloc((size_t)G4U * (2 * U_) * 2);
  unsigned short* W3  = (unsigned short*)alloc((size_t)G4U * (2 * U_) * 2);
  unsigned short* WT  = (unsigned short*)alloc((size_t)V_ * U_ * 2);
  unsigned short* INP = (unsigned short*)alloc((size_t)B_ * T_ * D0_ * 2);
  unsigned short* Yb  = (unsigned short*)alloc((size_t)B_ * T_ * U_ * 2);
  unsigned short* H   = (unsigned short*)alloc((size_t)6 * B_ * U_ * 2);
  float*          C   = (float*)alloc((size_t)3 * B_ * U_ * 4);

  unsigned short* h1[2] = { H + 0 * B_ * U_, H + 1 * B_ * U_ };
  unsigned short* h2[2] = { H + 2 * B_ * U_, H + 3 * B_ * U_ };
  unsigned short* h3[2] = { H + 4 * B_ * U_, H + 5 * B_ * U_ };
  float* c1 = C;
  float* c2 = C + B_ * U_;
  float* c3 = C + 2 * B_ * U_;

  float* out_y      = (float*)d_out;                     // softmax output [B,T,V]
  float* out_logits = out_y + (size_t)B_ * T_ * V_;      // logits        [B,T,V]

  // ---- prep: fused transposed bf16 weights, concat input, zero state ----
  {
    long tot = (long)G4U * (D0_ + U_);
    prep_weights<<<(unsigned)((tot + 255) / 256), 256, 0, stream>>>(k1, rk1, W1, D0_, U_, G4U);
  }
  {
    long tot = (long)G4U * (2 * U_);
    prep_weights<<<(unsigned)((tot + 255) / 256), 256, 0, stream>>>(k2, rk2, W2, U_, U_, G4U);
    prep_weights<<<(unsigned)((tot + 255) / 256), 256, 0, stream>>>(k3, rk3, W3, U_, U_, G4U);
  }
  {
    long tot = (long)V_ * U_;
    prep_weights<<<(unsigned)((tot + 255) / 256), 256, 0, stream>>>(w, nullptr, WT, U_, 0, V_);
  }
  {
    long tot = (long)B_ * T_ * D0_;
    prep_concat<<<(unsigned)((tot + 255) / 256), 256, 0, stream>>>(latent, x, INP);
  }
  prep_init<<<(6 * B_ * U_ + 255) / 256, 256, 0, stream>>>(H, C);

  // ---- recurrence: 3 layer-launches per timestep (launch order = sync) ----
  for (int t = 0; t < T_; ++t) {
    const int pi = t & 1, ni = pi ^ 1;
    lstm_cell<<<U_ / 16, 512, 0, stream>>>(
        INP + (size_t)t * D0_, (long)T_ * D0_, D0_,
        h1[pi], W1, D0_ + U_, b1, c1, h1[ni], nullptr, 0);
    lstm_cell<<<U_ / 16, 512, 0, stream>>>(
        h1[ni], (long)U_, U_,
        h2[pi], W2, 2 * U_, b2, c2, h2[ni], nullptr, 0);
    lstm_cell<<<U_ / 16, 512, 0, stream>>>(
        h2[ni], (long)U_, U_,
        h3[pi], W3, 2 * U_, b3, c3, h3[ni],
        Yb + (size_t)t * U_, (long)T_ * U_);
  }

  // ---- output projection + softmax ----
  proj_logits<<<dim3((B_ * T_) / 64, V_ / 32), 256, 0, stream>>>(Yb, WT, bb, out_logits);
  softmax512<<<B_ * T_, 256, 0, stream>>>(out_logits, out_y);
}
